// MultiHeadAttention_47055661695519
// MI455X (gfx1250) — compile-verified, matrix operand order checked by reading, not run on hardware
//
#include <hip/hip_runtime.h>
#include <hip/hip_bf16.h>

typedef _Float16 half_t;
typedef __attribute__((ext_vector_type(16))) _Float16 v16h;
typedef __attribute__((ext_vector_type(8)))  _Float16 v8h;
typedef __attribute__((ext_vector_type(8)))  float    v8f;
typedef __attribute__((ext_vector_type(4)))  int      v4i;

#define D_MODEL 1024
#define T_SEQ   4096
#define NHEAD   16
#define DK      64

// ---------------------------------------------------------------------------
// Optional CDNA5 async global->LDS path (ASYNCcnt).  Guarded so that if the
// toolchain lacks the builtins we fall back to plain load+ds_store staging.
// Probe-learned signature: (int4* global_src, int4* lds_dst, imm offset, imm cpol)
// with generic pointers (clang inserts the AS1/AS3 casts).
// ---------------------------------------------------------------------------
#if defined(__AMDGCN__) && __has_builtin(__builtin_amdgcn_global_load_async_to_lds_b128) && __has_builtin(__builtin_amdgcn_s_wait_asynccnt)
#define HAVE_ASYNC_LDS 1
#else
#define HAVE_ASYNC_LDS 0
#endif

#if HAVE_ASYNC_LDS
#define TO_GLB(p) ((v4i*)(p))
#define TO_LDS(p) ((v4i*)(p))
#endif

// ---------------------------------------------------------------------------
// WMMA helpers. Fragment layout per CDNA5 ISA (16-bit A 16x32 / mirrored B):
//   lane row = lane%16, k-base = (lane<16 ? 0 : 8), halves [0..7] at k-base,
//   halves [8..15] at k-base+16.  C/D: vgpr r -> row r + (lane<16?0:8).
// ---------------------------------------------------------------------------
__device__ __forceinline__ v16h frag_ld(const half_t* rowptr, int kb) {
    v8h lo = *(const v8h*)(rowptr + kb);
    v8h hi = *(const v8h*)(rowptr + kb + 16);
    v16h r;
#pragma unroll
    for (int i = 0; i < 8; ++i) { r[i] = lo[i]; r[i + 8] = hi[i]; }
    return r;
}

__device__ __forceinline__ v8f wmma16(v16h a, v16h b, v8f c) {
    return __builtin_amdgcn_wmma_f32_16x16x32_f16(
        /*neg_a=*/false, a, /*neg_b=*/false, b,
        /*c_mod=*/(short)0, c, /*reuse_a=*/false, /*reuse_b=*/false);
}

// ---------------------------------------------------------------------------
// fp32 -> fp16 conversion, 8 elements / thread / step (b128 loads, b128 store)
// ---------------------------------------------------------------------------
__global__ __launch_bounds__(256) void cvt_f32_to_f16_k(
    const float* __restrict__ s, half_t* __restrict__ d, int n) {
    const int stride = gridDim.x * blockDim.x;
    for (int idx = blockIdx.x * blockDim.x + threadIdx.x; idx * 8 < n; idx += stride) {
        const float4 a = ((const float4*)s)[idx * 2];
        const float4 b = ((const float4*)s)[idx * 2 + 1];
        v8h o;
        o[0] = (half_t)a.x; o[1] = (half_t)a.y; o[2] = (half_t)a.z; o[3] = (half_t)a.w;
        o[4] = (half_t)b.x; o[5] = (half_t)b.y; o[6] = (half_t)b.z; o[7] = (half_t)b.w;
        ((v8h*)d)[idx] = o;
    }
}

// ---------------------------------------------------------------------------
// NT GEMM: C[m,n] = sum_k A[m,k]*B[n,k]  (A: MxK row-major, B: NxK row-major)
// Block tile 128x128, BK=32, double-buffered LDS, async global->LDS staging.
// 8 waves in 4x2; each wave computes 32x64 (8 WMMA / k-step, 12 ds_b128).
// ---------------------------------------------------------------------------
#define GBM 128
#define GBN 128
#define GBK 32
#define LDT_P 40  // padded row stride in halfs (80B: 16B aligned, conflict-free)

#if HAVE_ASYNC_LDS
#define STAGE_GEMM(kt, buf) do {                                               \
    const half_t* ga = A + (size_t)(m0 + ldr) * K + (kt) + ldk;                \
    const half_t* gb = B + (size_t)(n0 + ldr) * K + (kt) + ldk;                \
    __builtin_amdgcn_global_load_async_to_lds_b128(                            \
        TO_GLB(ga),     TO_LDS(&sA[buf][ldr * LDT_P + ldk]),     0, 0);        \
    __builtin_amdgcn_global_load_async_to_lds_b128(                            \
        TO_GLB(ga + 8), TO_LDS(&sA[buf][ldr * LDT_P + ldk + 8]), 0, 0);        \
    __builtin_amdgcn_global_load_async_to_lds_b128(                            \
        TO_GLB(gb),     TO_LDS(&sB[buf][ldr * LDT_P + ldk]),     0, 0);        \
    __builtin_amdgcn_global_load_async_to_lds_b128(                            \
        TO_GLB(gb + 8), TO_LDS(&sB[buf][ldr * LDT_P + ldk + 8]), 0, 0);        \
} while (0)
#define STAGE_WAIT() __builtin_amdgcn_s_wait_asynccnt(0)
#else
#define STAGE_GEMM(kt, buf) do {                                               \
    const half_t* ga = A + (size_t)(m0 + ldr) * K + (kt) + ldk;                \
    const half_t* gb = B + (size_t)(n0 + ldr) * K + (kt) + ldk;                \
    v8h a0 = *(const v8h*)ga; v8h a1 = *(const v8h*)(ga + 8);                  \
    v8h b0 = *(const v8h*)gb; v8h b1 = *(const v8h*)(gb + 8);                  \
    *(v8h*)&sA[buf][ldr * LDT_P + ldk]     = a0;                               \
    *(v8h*)&sA[buf][ldr * LDT_P + ldk + 8] = a1;                               \
    *(v8h*)&sB[buf][ldr * LDT_P + ldk]     = b0;                               \
    *(v8h*)&sB[buf][ldr * LDT_P + ldk + 8] = b1;                               \
} while (0)
#define STAGE_WAIT() do {} while (0)
#endif

template <bool OUT_F32>
__global__ __launch_bounds__(256) void gemm_nt_wmma(
    const half_t* __restrict__ A, const half_t* __restrict__ B,
    void* __restrict__ Cv, int M, int N, int K) {
    __shared__ __align__(16) half_t sA[2][GBM * LDT_P];
    __shared__ __align__(16) half_t sB[2][GBN * LDT_P];

    const int tid  = threadIdx.x;
    const int lane = tid & 31;
    const int wave = tid >> 5;
    const int lr   = lane & 15;
    const int kb   = (lane < 16) ? 0 : 8;
    const int hi8  = (lane < 16) ? 0 : 8;
    const int m0   = blockIdx.y * GBM;
    const int n0   = blockIdx.x * GBN;
    const int mi0  = (wave & 3) * 32;   // wave row offset in block tile
    const int ni0  = (wave >> 2) * 64;  // wave col offset in block tile

    // cooperative loader: 256 threads, each stages 16 halfs of A and of B
    const int ldr = tid >> 1;           // 0..127 tile row
    const int ldk = (tid & 1) * 16;     // 0 / 16 k-offset

    v8f acc[2][4] = {};

    const int NT = K / GBK;
    STAGE_GEMM(0, 0);
    STAGE_WAIT();

    for (int t = 0; t < NT; ++t) {
        __syncthreads();
        const int cur = t & 1;
        if (t + 1 < NT) STAGE_GEMM((t + 1) * GBK, cur ^ 1);

        const half_t* pa = &sA[cur][0];
        const half_t* pb = &sB[cur][0];
        v16h af0 = frag_ld(pa + (mi0 + lr) * LDT_P, kb);
        v16h af1 = frag_ld(pa + (mi0 + 16 + lr) * LDT_P, kb);
#pragma unroll
        for (int nt = 0; nt < 4; ++nt) {
            v16h bf = frag_ld(pb + (ni0 + nt * 16 + lr) * LDT_P, kb);
            acc[0][nt] = wmma16(af0, bf, acc[0][nt]);
            acc[1][nt] = wmma16(af1, bf, acc[1][nt]);
        }
        if (t + 1 < NT) STAGE_WAIT();  // async staging overlapped the WMMAs
    }

#pragma unroll
    for (int mt = 0; mt < 2; ++mt) {
#pragma unroll
        for (int nt = 0; nt < 4; ++nt) {
            const int col = n0 + ni0 + nt * 16 + lr;
#pragma unroll
            for (int r = 0; r < 8; ++r) {
                const int row = m0 + mi0 + mt * 16 + r + hi8;
                if (OUT_F32)
                    ((float*)Cv)[(size_t)row * N + col] = acc[mt][nt][r];
                else
                    ((half_t*)Cv)[(size_t)row * N + col] = (half_t)acc[mt][nt][r];
            }
        }
    }
}

// ---------------------------------------------------------------------------
// Causal flash attention, one head x one 64-row query block per CTA.
// 8 waves; wave w owns tiles (mi = w/2, ni = (w&1)*2 + {0,1}).
// K tile staged async to LDS; softmax parallel over all 256 threads with
// wave32 shfl_xor reductions (4 lanes per row).
// ---------------------------------------------------------------------------
__global__ __launch_bounds__(256) void flash_attn_wmma(
    const half_t* __restrict__ Q, const half_t* __restrict__ Kp,
    const half_t* __restrict__ V, half_t* __restrict__ O) {
    __shared__ __align__(16) half_t sK [64 * 72];  // [key][d]
    __shared__ __align__(16) half_t sVt[64 * 72];  // [d][key]  (transposed V)
    __shared__ __align__(16) float  sS [64 * 68];  // fp32 scores
    __shared__ __align__(16) half_t sP [64 * 72];  // fp16 probabilities
    __shared__ float sM[64], sL[64], sAlpha[64];

    const int tid  = threadIdx.x;
    const int lane = tid & 31;
    const int wave = tid >> 5;
    const int lr   = lane & 15;
    const int kb   = (lane < 16) ? 0 : 8;
    const int hi8  = (lane < 16) ? 0 : 8;
    const int qb   = blockIdx.x;
    const int h    = blockIdx.y;
    const int mi   = wave >> 1;
    const int nib  = (wave & 1) * 2;

    // Q fragments held in registers across the whole key loop
    const half_t* qrow = Q + (size_t)(qb * 64 + mi * 16 + lr) * D_MODEL + h * DK;
    const v16h qf0 = frag_ld(qrow, kb);
    const v16h qf1 = frag_ld(qrow + 32, kb);

    if (tid < 64) { sM[tid] = -3.0e38f; sL[tid] = 0.0f; }

    v8f acc[2] = {};

    const int r4 = tid >> 2;        // 0..63 loader / softmax row
    const int c4 = (tid & 3) * 16;  // 0/16/32/48 column segment

    for (int kblk = 0; kblk <= qb; ++kblk) {
        // ---- stage K (row-major, async when available) and V (transposed) --
        {
            const half_t* g = Kp + (size_t)(kblk * 64 + r4) * D_MODEL + h * DK + c4;
#if HAVE_ASYNC_LDS
            __builtin_amdgcn_global_load_async_to_lds_b128(
                TO_GLB(g),     TO_LDS(sK + r4 * 72 + c4),     0, 0);
            __builtin_amdgcn_global_load_async_to_lds_b128(
                TO_GLB(g + 8), TO_LDS(sK + r4 * 72 + c4 + 8), 0, 0);
#else
            *(v8h*)(sK + r4 * 72 + c4)     = *(const v8h*)(g);
            *(v8h*)(sK + r4 * 72 + c4 + 8) = *(const v8h*)(g + 8);
#endif
        }
        {
            const half_t* g = V + (size_t)(kblk * 64 + r4) * D_MODEL + h * DK + c4;
            v8h v0 = *(const v8h*)(g);
            v8h v1 = *(const v8h*)(g + 8);
#pragma unroll
            for (int j = 0; j < 8; ++j) {
                sVt[(c4 + j) * 72 + r4]     = v0[j];
                sVt[(c4 + 8 + j) * 72 + r4] = v1[j];
            }
        }
#if HAVE_ASYNC_LDS
        __builtin_amdgcn_s_wait_asynccnt(0);
#endif
        __syncthreads();

        // ---- S = (Q K^T) / sqrt(dk), causal-masked, fp32 into LDS ---------
#pragma unroll
        for (int i = 0; i < 2; ++i) {
            const int ni = nib + i;
            v16h kf0 = frag_ld(sK + (ni * 16 + lr) * 72, kb);
            v16h kf1 = frag_ld(sK + (ni * 16 + lr) * 72 + 32, kb);
            v8f s = {};
            s = wmma16(qf0, kf0, s);
            s = wmma16(qf1, kf1, s);
            const int col_l = ni * 16 + lr;
            const int gcol  = kblk * 64 + col_l;
#pragma unroll
            for (int r = 0; r < 8; ++r) {
                const int row_l = mi * 16 + r + hi8;
                const int grow  = qb * 64 + row_l;
                float val = s[r] * 0.125f;               // 1/sqrt(64)
                if (gcol > grow) val = -3.0e38f;         // causal mask
                sS[row_l * 68 + col_l] = val;
            }
        }
        __syncthreads();

        // ---- fp32 online softmax: 4 lanes per row, shfl_xor reductions ----
        {
            const float  mOld = sM[r4];
            const float* srw  = sS + r4 * 68 + c4;
            float mx = mOld;
#pragma unroll
            for (int j = 0; j < 16; ++j) mx = fmaxf(mx, srw[j]);
            mx = fmaxf(mx, __shfl_xor(mx, 1));
            mx = fmaxf(mx, __shfl_xor(mx, 2));
            half_t* prw = sP + r4 * 72 + c4;
            float sum = 0.0f;
#pragma unroll
            for (int j = 0; j < 16; ++j) {
                const float p = __expf(srw[j] - mx);
                prw[j] = (half_t)p;
                sum += p;
            }
            sum += __shfl_xor(sum, 1);
            sum += __shfl_xor(sum, 2);
            if ((tid & 3) == 0) {
                const float alpha = __expf(mOld - mx);
                sL[r4]     = sL[r4] * alpha + sum;
                sM[r4]     = mx;
                sAlpha[r4] = alpha;
            }
        }
        __syncthreads();

        // ---- rescale accumulators, then O += P * V ------------------------
#pragma unroll
        for (int r = 0; r < 8; ++r) {
            const float a = sAlpha[mi * 16 + r + hi8];
            acc[0][r] *= a;
            acc[1][r] *= a;
        }
#pragma unroll
        for (int i = 0; i < 2; ++i) {
            const int ni = nib + i;
#pragma unroll
            for (int ks = 0; ks < 64; ks += 32) {
                v16h pf = frag_ld(sP  + (mi * 16 + lr) * 72 + ks, kb);
                v16h vf = frag_ld(sVt + (ni * 16 + lr) * 72 + ks, kb);
                acc[i] = wmma16(pf, vf, acc[i]);
            }
        }
        __syncthreads();
    }

    // ---- normalize and write (T, H*dk) fp16 -------------------------------
#pragma unroll
    for (int i = 0; i < 2; ++i) {
        const int col = h * DK + (nib + i) * 16 + lr;
#pragma unroll
        for (int r = 0; r < 8; ++r) {
            const int row_l = mi * 16 + r + hi8;
            const float inv = 1.0f / sL[row_l];
            O[(size_t)(qb * 64 + row_l) * D_MODEL + col] = (half_t)(acc[i][r] * inv);
        }
    }
}

// ---------------------------------------------------------------------------
// Launch
// ---------------------------------------------------------------------------
extern "C" void kernel_launch(void* const* d_in, const int* in_sizes, int n_in,
                              void* d_out, int out_size, void* d_ws, size_t ws_size,
                              hipStream_t stream) {
    (void)in_sizes; (void)n_in; (void)out_size; (void)ws_size;

    const float* x  = (const float*)d_in[0];
    const float* Wq = (const float*)d_in[1];
    const float* Wk = (const float*)d_in[2];
    const float* Wv = (const float*)d_in[3];
    const float* Wo = (const float*)d_in[4];
    float* out = (float*)d_out;

    char* ws = (char*)d_ws;
    const size_t XH = (size_t)T_SEQ * D_MODEL * sizeof(half_t);   // 8 MiB
    const size_t WH = (size_t)D_MODEL * D_MODEL * sizeof(half_t); // 2 MiB

    half_t* x16  = (half_t*)(ws);
    half_t* wq16 = (half_t*)(ws + XH);
    half_t* wk16 = (half_t*)(ws + XH + 1 * WH);
    half_t* wv16 = (half_t*)(ws + XH + 2 * WH);
    half_t* wo16 = (half_t*)(ws + XH + 3 * WH);
    half_t* q16  = (half_t*)(ws + XH + 4 * WH);
    half_t* k16  = q16 + (size_t)T_SEQ * D_MODEL;
    half_t* v16p = k16 + (size_t)T_SEQ * D_MODEL;
    half_t* ao16 = v16p + (size_t)T_SEQ * D_MODEL;

    const int nX = T_SEQ * D_MODEL;
    const int nW = D_MODEL * D_MODEL;
    cvt_f32_to_f16_k<<<256, 256, 0, stream>>>(x,  x16,  nX);
    cvt_f32_to_f16_k<<<128, 256, 0, stream>>>(Wq, wq16, nW);
    cvt_f32_to_f16_k<<<128, 256, 0, stream>>>(Wk, wk16, nW);
    cvt_f32_to_f16_k<<<128, 256, 0, stream>>>(Wv, wv16, nW);
    cvt_f32_to_f16_k<<<128, 256, 0, stream>>>(Wo, wo16, nW);

    dim3 gProj(D_MODEL / GBN, T_SEQ / GBM);  // (8, 32)
    gemm_nt_wmma<false><<<gProj, 256, 0, stream>>>(x16, wq16, (void*)q16,
                                                   T_SEQ, D_MODEL, D_MODEL);
    gemm_nt_wmma<false><<<gProj, 256, 0, stream>>>(x16, wk16, (void*)k16,
                                                   T_SEQ, D_MODEL, D_MODEL);
    gemm_nt_wmma<false><<<gProj, 256, 0, stream>>>(x16, wv16, (void*)v16p,
                                                   T_SEQ, D_MODEL, D_MODEL);

    flash_attn_wmma<<<dim3(T_SEQ / 64, NHEAD), 256, 0, stream>>>(q16, k16, v16p, ao16);

    gemm_nt_wmma<true><<<gProj, 256, 0, stream>>>(ao16, wo16, (void*)out,
                                                  T_SEQ, D_MODEL, D_MODEL);
}
